// StandardHyperbolicQuantizer_74569222193360
// MI455X (gfx1250) — matrix operand order repaired and checked by reading, not compiled, not run on hardware
//
#include <hip/hip_runtime.h>
#include <hip/hip_bf16.h>
#include <math.h>

typedef __bf16 v16bf __attribute__((ext_vector_type(16)));
typedef __bf16 v4bf  __attribute__((ext_vector_type(4)));
typedef float  v8f   __attribute__((ext_vector_type(8)));

#define N_PTS 8192
#define NE    8192
#define DDIM  256
#define NTILES (NE / 32)
#define EPS_  1e-7f

// output layout (floats): [loss][z_q 8192*256][perplexity][diversity][e_mean 8192]
#define OFF_ZQ    1
#define OFF_PERP  (1 + N_PTS * DDIM)
#define OFF_DIV   (OFF_PERP + 1)
#define OFF_EMEAN (OFF_DIV + 1)

#define WMMA_BF16(A, B, C) \
    __builtin_amdgcn_wmma_f32_16x16x32_bf16(false, (A), false, (B), (short)0, (C), false, false)

// workspace layout: int idx[8192] | float counts[8192] | float dist[8192]

__global__ __launch_bounds__(256) void vq_init_kernel(float* __restrict__ counts) {
    int i = blockIdx.x * 256 + threadIdx.x;
    if (i < NE) counts[i] = 0.0f;
}

// Fused (u*signs)@cbT argmin via bf16x3 WMMA. 64 blocks x 256 threads.
// Block tile: 128 rows (wave w -> rows w*16..w*16+15), loop codebook in 32-code tiles.
// - 4 independent WMMA accumulator chains (2 subtiles x even/odd K-chunks)
// - next tile's global fetch software-pipelined through registers so VMEM latency
//   overlaps the 48-WMMA compute block instead of sitting between the barriers.
__global__ __launch_bounds__(256) void vq_argmin_kernel(const float* __restrict__ u,
                                                        const float* __restrict__ cb,
                                                        int* __restrict__ wsIdx) {
    __shared__ __bf16 sHi[32 * DDIM];   // 16 KB
    __shared__ __bf16 sLo[32 * DDIM];   // 16 KB
    __shared__ float  redV[8 * 8 * 32]; // 8 KB
    __shared__ int    redI[8 * 8 * 32]; // 8 KB

    const int tid  = threadIdx.x;
    const int wv   = tid >> 5;
    const int ln   = tid & 31;
    const int half = ln >> 4;       // K-half selector
    const int lrow = ln & 15;       // A row / B column within 16
    const int m0   = blockIdx.x * 128;
    const int row  = m0 + wv * 16 + lrow;

    // ---- A fragments: u row, Minkowski sign on k==0, bf16 hi/lo split, kept in VGPRs ----
    v16bf ahi[8], alo[8];
#pragma unroll
    for (int c = 0; c < 8; ++c) {
        const int base = 32 * c + 16 * half;  // lane holds K in [base,base+16)
        const float4 fa = *(const float4*)(u + row * DDIM + base);
        const float4 fb = *(const float4*)(u + row * DDIM + base + 4);
        const float4 fc = *(const float4*)(u + row * DDIM + base + 8);
        const float4 fd = *(const float4*)(u + row * DDIM + base + 12);
        float vals[16] = {fa.x, fa.y, fa.z, fa.w, fb.x, fb.y, fb.z, fb.w,
                          fc.x, fc.y, fc.z, fc.w, fd.x, fd.y, fd.z, fd.w};
        if (c == 0 && half == 0) vals[0] = -vals[0];  // time component sign
#pragma unroll
        for (int e = 0; e < 16; ++e) {
            __bf16 h = (__bf16)vals[e];
            ahi[c][e] = h;
            alo[c][e] = (__bf16)(vals[e] - (float)h);
        }
    }

    float best[8];
    int   bidx[8];
#pragma unroll
    for (int j = 0; j < 8; ++j) { best[j] = 3.4e38f; bidx[j] = 0; }

    // ---- pipeline prologue: fetch tile 0 into registers ----
    float4 pf[8];
#pragma unroll
    for (int i = 0; i < 8; ++i)
        pf[i] = *(const float4*)(cb + i * 1024 + tid * 4);

    for (int nt = 0; nt < NTILES; ++nt) {
        __syncthreads();  // previous iteration done reading LDS
        // convert staged tile nt (registers) -> bf16 hi/lo in LDS
#pragma unroll
        for (int i = 0; i < 8; ++i) {
            const int f4 = i * 1024 + tid * 4;  // flat float index within 32x256 tile
            float xs[4] = {pf[i].x, pf[i].y, pf[i].z, pf[i].w};
            v4bf h, l;
#pragma unroll
            for (int e = 0; e < 4; ++e) {
                __bf16 hh = (__bf16)xs[e];
                h[e] = hh;
                l[e] = (__bf16)(xs[e] - (float)hh);
            }
            *(v4bf*)(sHi + f4) = h;
            *(v4bf*)(sLo + f4) = l;
        }
        __syncthreads();

        // issue next tile's global loads now; latency hides under the WMMA block
        if (nt + 1 < NTILES) {
#pragma unroll
            for (int i = 0; i < 8; ++i)
                pf[i] = *(const float4*)(cb + (nt + 1) * 32 * DDIM + i * 1024 + tid * 4);
        }
        if (nt + 2 < NTILES)
            __builtin_prefetch(cb + (nt + 2) * 32 * DDIM + tid * 32, 0, 3);

        // 4 independent accumulator chains: {subtile0, subtile1} x {even c, odd c}
        v8f a0e = {0.f,0.f,0.f,0.f,0.f,0.f,0.f,0.f};
        v8f a1e = {0.f,0.f,0.f,0.f,0.f,0.f,0.f,0.f};
        v8f a0o = {0.f,0.f,0.f,0.f,0.f,0.f,0.f,0.f};
        v8f a1o = {0.f,0.f,0.f,0.f,0.f,0.f,0.f,0.f};
#pragma unroll
        for (int c = 0; c < 8; ++c) {
            // B layout: lanes0-15 hold K=kc..kc+15, lanes16-31 hold K=kc+16..kc+31
            const int off0 = lrow * DDIM + 32 * c + 16 * half;          // subtile 0
            const int off1 = (16 + lrow) * DDIM + 32 * c + 16 * half;   // subtile 1
            v16bf b0h = *(const v16bf*)(sHi + off0);
            v16bf b0l = *(const v16bf*)(sLo + off0);
            v16bf b1h = *(const v16bf*)(sHi + off1);
            v16bf b1l = *(const v16bf*)(sLo + off1);
            if (c & 1) {
                a0o = WMMA_BF16(alo[c], b0h, a0o);
                a1o = WMMA_BF16(alo[c], b1h, a1o);
                a0o = WMMA_BF16(ahi[c], b0l, a0o);
                a1o = WMMA_BF16(ahi[c], b1l, a1o);
                a0o = WMMA_BF16(ahi[c], b0h, a0o);
                a1o = WMMA_BF16(ahi[c], b1h, a1o);
            } else {
                a0e = WMMA_BF16(alo[c], b0h, a0e);
                a1e = WMMA_BF16(alo[c], b1h, a1e);
                a0e = WMMA_BF16(ahi[c], b0l, a0e);
                a1e = WMMA_BF16(ahi[c], b1l, a1e);
                a0e = WMMA_BF16(ahi[c], b0h, a0e);
                a1e = WMMA_BF16(ahi[c], b1h, a1e);
            }
        }
        const v8f k0 = a0e + a0o;
        const v8f k1 = a1e + a1o;

        // subtile 0 first, then subtile 1: preserves first-min index semantics
        const int n0 = nt * 32 + lrow;
        const int n1 = n0 + 16;
#pragma unroll
        for (int j = 0; j < 8; ++j) {
            const float m = fmaxf(-k0[j], 1.0f + EPS_);  // arccosh argument; monotone
            if (m < best[j]) { best[j] = m; bidx[j] = n0; }
        }
#pragma unroll
        for (int j = 0; j < 8; ++j) {
            const float m = fmaxf(-k1[j], 1.0f + EPS_);
            if (m < best[j]) { best[j] = m; bidx[j] = n1; }
        }
    }

    // ---- cross-lane argmin: C-tile row M = j + 8*half lives in 16 lanes of one half ----
    __syncthreads();
#pragma unroll
    for (int j = 0; j < 8; ++j) {
        redV[(wv * 8 + j) * 32 + ln] = best[j];
        redI[(wv * 8 + j) * 32 + ln] = bidx[j];
    }
    __syncthreads();
    if (tid < 128) {
        const int r = tid;
        const int w = r >> 4, rw = r & 15, h = rw >> 3, j = rw & 7;
        const int base = (w * 8 + j) * 32 + h * 16;
        float bv = 3.4e38f; int bi = 0;
#pragma unroll
        for (int l = 0; l < 16; ++l) {
            float v = redV[base + l];
            int   id = redI[base + l];
            if (v < bv || (v == bv && id < bi)) { bv = v; bi = id; }
        }
        wsIdx[m0 + r] = bi;
    }
}

// One block per point: z_q gather, exact fp32 distance, count histogram.
__global__ __launch_bounds__(256) void vq_gather_kernel(const float* __restrict__ u,
                                                        const float* __restrict__ cb,
                                                        const int* __restrict__ wsIdx,
                                                        float* __restrict__ counts,
                                                        float* __restrict__ wsDist,
                                                        float* __restrict__ out_zq) {
    __shared__ float red[256];
    const int i = blockIdx.x;
    const int t = threadIdx.x;
    const int idx = wsIdx[i];
    const float cv = cb[idx * DDIM + t];
    out_zq[i * DDIM + t] = cv;
    float p = u[i * DDIM + t] * cv;
    if (t == 0) p = -p;                 // Minkowski sign
    red[t] = p;
    __syncthreads();
#pragma unroll
    for (int sft = 128; sft > 0; sft >>= 1) {
        if (t < sft) red[t] += red[t + sft];
        __syncthreads();
    }
    if (t == 0) {
        wsDist[i] = acoshf(fmaxf(-red[0], 1.0f + EPS_));
        atomicAdd(&counts[idx], 1.0f);  // exact integer adds -> order independent
    }
}

__global__ __launch_bounds__(256) void vq_finalize_kernel(const float* __restrict__ counts,
                                                          const float* __restrict__ wsDist,
                                                          float* __restrict__ out) {
    __shared__ float red[256];
    const int t = threadIdx.x;
    float ent = 0.0f, dsum = 0.0f;
    for (int k = t; k < NE; k += 256) {
        float e = counts[k] * (1.0f / (float)N_PTS);
        out[OFF_EMEAN + k] = e;
        ent -= e * logf(e + 1e-10f);
        dsum += wsDist[k];
    }
    red[t] = ent;
    __syncthreads();
    for (int sft = 128; sft > 0; sft >>= 1) {
        if (t < sft) red[t] += red[t + sft];
        __syncthreads();
    }
    float entropy = red[0];
    __syncthreads();
    red[t] = dsum;
    __syncthreads();
    for (int sft = 128; sft > 0; sft >>= 1) {
        if (t < sft) red[t] += red[t + sft];
        __syncthreads();
    }
    if (t == 0) {
        out[0]        = red[0] * (1.0f / (float)N_PTS) * 1.25f;  // (1 + BETA) * mean dist
        out[OFF_PERP] = expf(entropy);
        out[OFF_DIV]  = entropy;
    }
}

extern "C" void kernel_launch(void* const* d_in, const int* in_sizes, int n_in,
                              void* d_out, int out_size, void* d_ws, size_t ws_size,
                              hipStream_t stream) {
    const float* u  = (const float*)d_in[0];
    const float* cb = (const float*)d_in[1];
    float* out = (float*)d_out;

    int*   wsIdx  = (int*)d_ws;
    float* counts = (float*)d_ws + NE;
    float* wsDist = (float*)d_ws + 2 * NE;

    vq_init_kernel<<<(NE + 255) / 256, 256, 0, stream>>>(counts);
    vq_argmin_kernel<<<N_PTS / 128, 256, 0, stream>>>(u, cb, wsIdx);
    vq_gather_kernel<<<N_PTS, 256, 0, stream>>>(u, cb, wsIdx, counts, wsDist, out + OFF_ZQ);
    vq_finalize_kernel<<<1, 256, 0, stream>>>(counts, wsDist, out);
}
